// AttentionFlow_69363721830957
// MI455X (gfx1250) — matrix operand order repaired and checked by reading, not compiled, not run on hardware
//
#include <hip/hip_runtime.h>
#include <hip/hip_bf16.h>

// ---------------------------------------------------------------------------
// BiDAF attention flow for MI455X (gfx1250, wave32, WMMA).
// B=16, C_LEN=2048, Q_LEN=512, D=256.
// ---------------------------------------------------------------------------

#define BATCH 16
#define CLEN  2048
#define QLEN  512
#define DIM   256

typedef __attribute__((ext_vector_type(16))) __bf16 bf16x16;
typedef __attribute__((ext_vector_type(8)))  __bf16 bf16x8;
typedef __attribute__((ext_vector_type(8)))  float  f32x8;

union AOp { bf16x16 v; bf16x8 h[2]; };

// ---------------------------------------------------------------------------
// Prep 1: cb[b,i,d] = bf16(c * w_cq)   (WMMA A operand for the S matmul)
//         sc[b,i]  = c_row . w_c + b_c
// One wave per row, 8 rows per 256-thread block.
// ---------------------------------------------------------------------------
__global__ void __launch_bounds__(256)
prep_c_kernel(const float* __restrict__ c, const float* __restrict__ w_c,
              const float* __restrict__ b_c, const float* __restrict__ w_cq,
              __bf16* __restrict__ cb, float* __restrict__ sc) {
    const int wave = threadIdx.x >> 5;
    const int lane = threadIdx.x & 31;
    const int row  = blockIdx.x * 8 + wave;          // 0 .. B*CLEN-1
    const float* cr  = c  + (size_t)row * DIM;
    __bf16*      cbr = cb + (size_t)row * DIM;
    float acc = 0.f;
#pragma unroll
    for (int k = 0; k < 8; ++k) {
        const int d = lane + 32 * k;
        const float v = cr[d];
        cbr[d] = (__bf16)(v * w_cq[d]);
        acc += v * w_c[d];
    }
#pragma unroll
    for (int off = 16; off > 0; off >>= 1) acc += __shfl_xor(acc, off, 32);
    if (lane == 0) sc[row] = acc + b_c[0];
}

// ---------------------------------------------------------------------------
// Prep 2: qb[b,j,d]  = bf16(q)         (B operand for S matmul: contiguous d)
//         qTb[b,d,j] = bf16(q)         (B operand for P@Q matmul: contiguous j)
//         sq[b,j]    = q_row . w_q + b_q
// ---------------------------------------------------------------------------
__global__ void __launch_bounds__(256)
prep_q_kernel(const float* __restrict__ q, const float* __restrict__ w_q,
              const float* __restrict__ b_q,
              __bf16* __restrict__ qb, __bf16* __restrict__ qTb,
              float* __restrict__ sq) {
    const int wave = threadIdx.x >> 5;
    const int lane = threadIdx.x & 31;
    const int row  = blockIdx.x * 8 + wave;          // 0 .. B*QLEN-1
    const int b = row >> 9;                          // QLEN = 512
    const int j = row & (QLEN - 1);
    const float* qr  = q  + (size_t)row * DIM;
    __bf16*      qbr = qb + (size_t)row * DIM;
    float acc = 0.f;
#pragma unroll
    for (int k = 0; k < 8; ++k) {
        const int d = lane + 32 * k;
        const float v = qr[d];
        const __bf16 bv = (__bf16)v;
        qbr[d] = bv;
        qTb[((size_t)(b * DIM + d)) * QLEN + j] = bv;
        acc += v * w_q[d];
    }
#pragma unroll
    for (int off = 16; off > 0; off >>= 1) acc += __shfl_xor(acc, off, 32);
    if (lane == 0) sq[row] = acc + b_q[0];
}

// ---------------------------------------------------------------------------
// Flash attention kernel: one wave (32 threads) per 16 context rows.
//   S tile (16 x 512) -> LDS (fp32) -> exact row softmax -> P (bf16) in LDS
//   -> c2q = P @ q via WMMA with qTb as B operand.
// Row max (incl. sc[i] + b_cq) written out for the q2c pass.
// Static LDS = 16*528*4 + 16*528*2 + 64 = 50,752 B.
// ---------------------------------------------------------------------------
#define SSTR 528   // padded row stride (multiple of 8 -> 16B-aligned bf16 rows)

__global__ void __launch_bounds__(32)
attn_flash_kernel(const __bf16* __restrict__ cb, const __bf16* __restrict__ qb,
                  const __bf16* __restrict__ qTb,
                  const float* __restrict__ sc, const float* __restrict__ sq,
                  const float* __restrict__ b_cq,
                  float* __restrict__ c2q, float* __restrict__ m_out) {
    __shared__ __align__(16) float  S[16][SSTR];
    __shared__ __align__(16) __bf16 P[16][SSTR];
    __shared__ float invs[16];

    const int lane   = threadIdx.x & 31;
    const int lane16 = lane & 15;
    const int half   = lane >> 4;
    const int b      = blockIdx.y;
    const int i0     = blockIdx.x * 16;          // first context row of tile
    const int row_i  = i0 + lane16;              // A-operand row for this lane

    // ---- A operand: 8 K-chunks of (c*w_cq) rows, bf16.
    // A as loaded here (contiguous k, matched by B): lane L = row L%16; lanes<16 hold K 0-15,
    // lanes>=16 hold K 16-31 (two 16B chunks per lane).
    AOp A[8];
    const __bf16* cbr = cb + ((size_t)(b * CLEN + row_i)) * DIM;
#pragma unroll
    for (int kt = 0; kt < 8; ++kt) {
        const int d0 = kt * 32 + half * 16;
        A[kt].h[0] = *(const bf16x8*)(cbr + d0);
        A[kt].h[1] = *(const bf16x8*)(cbr + d0 + 8);
    }

    // ---- S = (c*w_cq) @ q^T + sq[j], striped into LDS.
    for (int jt = 0; jt < QLEN / 16; ++jt) {
        const int j = jt * 16 + lane16;          // B column = lane16
        f32x8 acc = {};
        const __bf16* qrow = qb + ((size_t)(b * QLEN + j)) * DIM;
#pragma unroll
        for (int kt = 0; kt < 8; ++kt) {
            // B as loaded, same k order as A: lanes<16 hold K 0-15, lanes>=16 hold K 16-31 (32B/lane)
            bf16x16 bv = *(const bf16x16*)(qrow + kt * 32 + half * 16);
            acc = __builtin_amdgcn_wmma_f32_16x16x32_bf16(
                false, A[kt].v, false, bv, (short)0, acc, false, false);
        }
        const float sqv = sq[b * QLEN + j];
#pragma unroll
        for (int r = 0; r < 8; ++r)              // C layout: row = r + half*8
            S[r + half * 8][j] = acc[r] + sqv;
    }
    __syncthreads();

    // ---- Exact softmax over j. Two lanes per row, 256 columns each.
    const int srow = lane >> 1;
    const int part = lane & 1;
    const float* Srow = &S[srow][part * 256];
    float mx = -3.4e38f;
    for (int jj = 0; jj < 256; ++jj) mx = fmaxf(mx, Srow[jj]);
    mx = fmaxf(mx, __shfl_xor(mx, 1, 32));

    __bf16* Prow = &P[srow][part * 256];
    float sum = 0.f;
    for (int jj = 0; jj < 256; ++jj) {
        const float e = __expf(Srow[jj] - mx);
        Prow[jj] = (__bf16)e;
        sum += e;
    }
    sum += __shfl_xor(sum, 1, 32);
    if (part == 0) {
        invs[srow] = 1.f / sum;
        const int gi = b * CLEN + i0 + srow;
        m_out[gi] = mx + sc[gi] + b_cq[0];       // full s row-max for q2c pass
    }
    __syncthreads();

    // ---- c2q = softmax(S) @ q  via WMMA, B operand from transposed qTb.
    float inv[8];
#pragma unroll
    for (int r = 0; r < 8; ++r) inv[r] = invs[half * 8 + r];

    const __bf16* qt = qTb + (size_t)b * DIM * QLEN;
    for (int dt = 0; dt < DIM / 16; ++dt) {
        f32x8 acc = {};
        const __bf16* qtd = qt + (size_t)(dt * 16 + lane16) * QLEN;
#pragma unroll
        for (int kc = 0; kc < QLEN / 32; ++kc) {
            AOp pa;
            const int j0 = kc * 32 + half * 16;
            pa.h[0] = *(const bf16x8*)(&P[lane16][j0]);
            pa.h[1] = *(const bf16x8*)(&P[lane16][j0 + 8]);
            bf16x16 bv = *(const bf16x16*)(qtd + kc * 32 + half * 16);
            acc = __builtin_amdgcn_wmma_f32_16x16x32_bf16(
                false, pa.v, false, bv, (short)0, acc, false, false);
        }
#pragma unroll
        for (int r = 0; r < 8; ++r)
            c2q[((size_t)(b * CLEN + i0 + half * 8 + r)) * DIM + dt * 16 + lane16] =
                acc[r] * inv[r];
    }
}

// ---------------------------------------------------------------------------
// q2c: softmax over i of m[b,:], then q2c[b,d] = sum_i batt[i] * c[b,i,d].
// One 256-thread block per batch; thread t owns output feature d = t.
// ---------------------------------------------------------------------------
__global__ void __launch_bounds__(256)
q2c_kernel(const float* __restrict__ c, const float* __restrict__ m,
           float* __restrict__ q2c) {
    __shared__ float batt[CLEN];
    __shared__ float red[256];
    const int b = blockIdx.x, tid = threadIdx.x;

    float mx = -3.4e38f;
    for (int i = tid; i < CLEN; i += 256) {
        const float v = m[b * CLEN + i];
        batt[i] = v;
        mx = fmaxf(mx, v);
    }
    red[tid] = mx;
    __syncthreads();
    for (int s = 128; s > 0; s >>= 1) {
        if (tid < s) red[tid] = fmaxf(red[tid], red[tid + s]);
        __syncthreads();
    }
    mx = red[0];
    __syncthreads();

    float sum = 0.f;
    for (int i = tid; i < CLEN; i += 256) {
        const float e = __expf(batt[i] - mx);
        batt[i] = e;
        sum += e;
    }
    red[tid] = sum;
    __syncthreads();
    for (int s = 128; s > 0; s >>= 1) {
        if (tid < s) red[tid] += red[tid + s];
        __syncthreads();
    }
    const float invS = 1.f / red[0];

    float acc = 0.f;
    const float* cp = c + (size_t)b * CLEN * DIM + tid;
    for (int i = 0; i < CLEN; ++i) acc += batt[i] * cp[(size_t)i * DIM];
    q2c[b * DIM + tid] = acc * invS;
}

// ---------------------------------------------------------------------------
// Final assembly: x = concat([c, c2q, c*c2q, c*q2c], -1)
// ---------------------------------------------------------------------------
__global__ void __launch_bounds__(256)
finalize_kernel(const float* __restrict__ c, const float* __restrict__ c2q,
                const float* __restrict__ q2c, float* __restrict__ out) {
    const size_t idx = (size_t)blockIdx.x * 256 + threadIdx.x; // (b*CLEN+i)*DIM+d
    const int d = (int)(idx & (DIM - 1));
    const size_t bi = idx >> 8;                                // b*CLEN + i
    const int b = (int)(bi >> 11);                             // CLEN = 2048
    const float cv = c[idx];
    const float av = c2q[idx];
    const float gv = q2c[b * DIM + d];
    float* o = out + bi * (size_t)(4 * DIM) + d;
    o[0 * DIM] = cv;
    o[1 * DIM] = av;
    o[2 * DIM] = cv * av;
    o[3 * DIM] = cv * gv;
}

// ---------------------------------------------------------------------------
// Launch
// ---------------------------------------------------------------------------
extern "C" void kernel_launch(void* const* d_in, const int* in_sizes, int n_in,
                              void* d_out, int out_size, void* d_ws, size_t ws_size,
                              hipStream_t stream) {
    const float* c    = (const float*)d_in[0];
    const float* q    = (const float*)d_in[1];
    const float* w_c  = (const float*)d_in[2];
    const float* b_c  = (const float*)d_in[3];
    const float* w_q  = (const float*)d_in[4];
    const float* b_q  = (const float*)d_in[5];
    const float* w_cq = (const float*)d_in[6];
    const float* b_cq = (const float*)d_in[7];
    float* out = (float*)d_out;
    char*  ws  = (char*)d_ws;

    // Workspace partition (bytes):
    constexpr size_t CB_OFF  = 0;                               // 16*2048*256*2 = 16 MiB
    constexpr size_t QB_OFF  = CB_OFF  + (size_t)BATCH * CLEN * DIM * 2;
    constexpr size_t QTB_OFF = QB_OFF  + (size_t)BATCH * QLEN * DIM * 2;
    constexpr size_t SC_OFF  = QTB_OFF + (size_t)BATCH * QLEN * DIM * 2;
    constexpr size_t SQ_OFF  = SC_OFF  + (size_t)BATCH * CLEN * 4;
    constexpr size_t M_OFF   = SQ_OFF  + (size_t)BATCH * QLEN * 4;
    constexpr size_t C2Q_OFF = M_OFF   + (size_t)BATCH * CLEN * 4;
    constexpr size_t Q2C_OFF = C2Q_OFF + (size_t)BATCH * CLEN * DIM * 4;

    __bf16* cb   = (__bf16*)(ws + CB_OFF);
    __bf16* qb   = (__bf16*)(ws + QB_OFF);
    __bf16* qTb  = (__bf16*)(ws + QTB_OFF);
    float*  sc   = (float*)(ws + SC_OFF);
    float*  sq   = (float*)(ws + SQ_OFF);
    float*  mrow = (float*)(ws + M_OFF);
    float*  c2q  = (float*)(ws + C2Q_OFF);
    float*  q2c  = (float*)(ws + Q2C_OFF);

    prep_c_kernel<<<BATCH * CLEN / 8, 256, 0, stream>>>(c, w_c, b_c, w_cq, cb, sc);
    prep_q_kernel<<<BATCH * QLEN / 8, 256, 0, stream>>>(q, w_q, b_q, qb, qTb, sq);

    dim3 grid(CLEN / 16, BATCH);
    attn_flash_kernel<<<grid, 32, 0, stream>>>(cb, qb, qTb, sc, sq, b_cq, c2q, mrow);

    q2c_kernel<<<BATCH, 256, 0, stream>>>(c, mrow, q2c);

    finalize_kernel<<<(BATCH * CLEN * DIM) / 256, 256, 0, stream>>>(c, c2q, q2c, out);
}
